// Attn_Module_39883066310716
// MI455X (gfx1250) — matrix operand order repaired and checked by reading, not compiled, not run on hardware
//
#include <hip/hip_runtime.h>

typedef _Float16 v16h __attribute__((ext_vector_type(16)));
typedef _Float16 v8h  __attribute__((ext_vector_type(8)));
typedef float    v8f  __attribute__((ext_vector_type(8)));
typedef int      v4i  __attribute__((vector_size(16)));

#define WMMA_F16(A, B, C) \
    __builtin_amdgcn_wmma_f32_16x16x32_f16(false, (A), false, (B), (short)0, (C), false, false)

// ---- CDNA5 async global->LDS copy (guarded; falls back to plain LDS store) ----
#if defined(__has_builtin)
#if __has_builtin(__builtin_amdgcn_global_load_async_to_lds_b128)
#define HAVE_ASYNC 1
#endif
#endif
#ifndef HAVE_ASYNC
#define HAVE_ASYNC 0
#endif

#if HAVE_ASYNC
typedef __attribute__((address_space(1))) v4i* g_v4i_p;   // global int4*
typedef __attribute__((address_space(3))) v4i* l_v4i_p;   // LDS int4*
#endif

__device__ __forceinline__ void cp16_g2l(const _Float16* g, _Float16* l) {
#if HAVE_ASYNC
    __builtin_amdgcn_global_load_async_to_lds_b128(
        (g_v4i_p)(unsigned long long)(__UINTPTR_TYPE__)g,
        (l_v4i_p)(unsigned int)(__UINTPTR_TYPE__)l,
        0, 0);
#else
    *(v8h*)l = *(const v8h*)g;
#endif
}

__device__ __forceinline__ void wait_async() {
#if HAVE_ASYNC
#if __has_builtin(__builtin_amdgcn_s_wait_asynccnt)
    __builtin_amdgcn_s_wait_asynccnt(0);
#else
    asm volatile("s_wait_asynccnt 0" ::: "memory");
#endif
#endif
}

// ---------------------------------------------------------------------------
// Kernel 1: v = Wv @ x ; q = Wq @ v ; k = Wk @ v
// grid = B(8) * (L/128)(16) = 128 blocks, 256 threads (8 waves)
// Outputs: qT,kT as [B][L][64] f16 ; vR as [B][64][L] f16 ; out[ch 64..127] f32
// ---------------------------------------------------------------------------
__global__ __launch_bounds__(256) void vqk_kernel(
    const float* __restrict__ x,
    const float* __restrict__ Wv, const float* __restrict__ bv,
    const float* __restrict__ Wq, const float* __restrict__ bq,
    const float* __restrict__ Wk, const float* __restrict__ bk,
    _Float16* __restrict__ qT, _Float16* __restrict__ kT,
    _Float16* __restrict__ vR, float* __restrict__ out)
{
    __shared__ _Float16 sXT[128][40];   // x chunk, transposed: [n][k], padded
    __shared__ _Float16 sVT[128][72];   // v tile, transposed: [n][c], 16B-aligned rows

    const int b    = blockIdx.x >> 4;
    const int l0   = (blockIdx.x & 15) * 128;
    const int tid  = threadIdx.x;
    const int lane = tid & 31;
    const int w    = tid >> 5;
    const int m0   = (w & 3) * 16;      // output-channel tile base
    const int grp  = w >> 2;            // column-tile group (0..1)
    const int lrow = lane & 15;
    const int half = lane >> 4;         // 0: lanes 0-15, 1: lanes 16-31

    v8f acc[4] = {};

    // ---- v = Wv @ x, K = 512 in chunks of 32 ----
    for (int kc = 0; kc < 512; kc += 32) {
        {   // cooperative: thread owns one x row segment (contiguous 64B load),
            // scatter-transpose into LDS
            const int k  = tid >> 3;            // 0..31 (k-row in chunk)
            const int n0 = (tid & 7) * 16;      // column segment
            const float* xp = x + ((size_t)b * 512 + kc + k) * 2048 + l0 + n0;
#pragma unroll
            for (int i = 0; i < 16; ++i)
                sXT[n0 + i][k] = (_Float16)xp[i];
        }
        __syncthreads();

        // A fragment: Wv rows m0..m0+15, K-chunk kc (contiguous 32B runs)
        v16h a;
        {
            const int k0 = half ? 8 : 0;
            const float* wp = Wv + (size_t)(m0 + lrow) * 512 + kc;
#pragma unroll
            for (int i = 0; i < 8; ++i) {
                a[i]     = (_Float16)wp[k0 + i];
                a[8 + i] = (_Float16)wp[16 + k0 + i];
            }
        }
#pragma unroll
        for (int t = 0; t < 4; ++t) {
            const int n  = grp * 64 + t * 16 + lrow;
            const int kb = half ? 8 : 0;
            v16h bf;
#pragma unroll
            for (int i = 0; i < 16; ++i) bf[i] = sXT[n][kb + (i & 7) + ((i >> 3) << 4)];
            acc[t] = WMMA_F16(a, bf, acc[t]);
        }
        __syncthreads();
    }

    // ---- write v (+bias) into sVT transposed: lane owns fixed col n, 8 rows ----
#pragma unroll
    for (int t = 0; t < 4; ++t) {
        const int n = grp * 64 + t * 16 + lrow;
        v8h pk;
#pragma unroll
        for (int r = 0; r < 8; ++r)
            pk[r] = (_Float16)(acc[t][r] + bv[m0 + half * 8 + r]);
        *(v8h*)&sVT[n][m0 + half * 8] = pk;
    }
    __syncthreads();

    // ---- q = Wq @ v, k = Wk @ v (K = 64, two chunks of 32) ----
    v8f qacc[4] = {}, kacc[4] = {};
#pragma unroll
    for (int kc = 0; kc < 64; kc += 32) {
        v16h aq, ak;
        const int k0 = half ? 8 : 0;
        const float* wqp = Wq + (size_t)(m0 + lrow) * 64 + kc;
        const float* wkp = Wk + (size_t)(m0 + lrow) * 64 + kc;
#pragma unroll
        for (int i = 0; i < 8; ++i) {
            aq[i]     = (_Float16)wqp[k0 + i];
            aq[8 + i] = (_Float16)wqp[16 + k0 + i];
            ak[i]     = (_Float16)wkp[k0 + i];
            ak[8 + i] = (_Float16)wkp[16 + k0 + i];
        }
#pragma unroll
        for (int t = 0; t < 4; ++t) {
            const int n  = grp * 64 + t * 16 + lrow;
            const int kb = half ? 8 : 0;
            v16h bf;
#pragma unroll
            for (int i = 0; i < 16; ++i) bf[i] = sVT[n][kc + kb + (i & 7) + ((i >> 3) << 4)];
            qacc[t] = WMMA_F16(aq, bf, qacc[t]);
            kacc[t] = WMMA_F16(ak, bf, kacc[t]);
        }
    }

    // ---- store qT, kT as [B][L][64] f16 (one 16B store per tile per lane) ----
#pragma unroll
    for (int t = 0; t < 4; ++t) {
        const int l = l0 + grp * 64 + t * 16 + lrow;
        v8h pq, pk2;
#pragma unroll
        for (int r = 0; r < 8; ++r) {
            pq[r]  = (_Float16)(qacc[t][r] + bq[m0 + half * 8 + r]);
            pk2[r] = (_Float16)(kacc[t][r] + bk[m0 + half * 8 + r]);
        }
        const size_t base = ((size_t)b * 2048 + l) * 64 + m0 + half * 8;
        *(v8h*)(qT + base) = pq;
        *(v8h*)(kT + base) = pk2;
    }

    // ---- cooperative: v row-major f16 to ws + f32 to out channels 64..127 ----
    {
        const int n  = tid >> 1;            // 0..127
        const int c0 = (tid & 1) * 32;      // 0 or 32
#pragma unroll
        for (int i = 0; i < 32; ++i) {
            const float vv = (float)sVT[n][c0 + i];
            vR[((size_t)b * 64 + c0 + i) * 2048 + l0 + n]        = (_Float16)vv;
            out[((size_t)b * 128 + 64 + c0 + i) * 2048 + l0 + n] = vv;
        }
    }
}

// ---------------------------------------------------------------------------
// Kernel 2: flash attention. One 16-row l-tile per wave; the 8 waves of a
// block share one batch and stream the same k/v chunks, staged into LDS via
// double-buffered async global->LDS DMA.
// grid = 128 blocks * 8 waves = 1024 tiles = B(8) * L/16(128)
// ---------------------------------------------------------------------------
__global__ __launch_bounds__(256) void attn_kernel(
    const _Float16* __restrict__ qT, const _Float16* __restrict__ kT,
    const _Float16* __restrict__ vR, const float* __restrict__ gptr,
    float* __restrict__ out)
{
    __shared__ _Float16 sK[2][32][72];   // k chunk: [j][c], padded rows (16B aligned)
    __shared__ _Float16 sV[2][64][40];   // v chunk: [c][j], padded rows (16B aligned)
    __shared__ _Float16 sP[8][16][32];   // wave-private P staging (D->A relayout)

    const int tid  = threadIdx.x;
    const int lane = tid & 31;
    const int w    = tid >> 5;
    const int wid  = blockIdx.x * 8 + w;
    const int b    = wid >> 7;           // block-uniform
    const int l0   = (wid & 127) * 16;
    const int lrow = lane & 15;
    const int half = lane >> 4;
    const float gamma = gptr[0];

    // per-thread staging slots: 16B per thread per buffer
    const int kj   = tid >> 3, kseg = tid & 7;   // sK: 32 rows x 8 segs
    const int vc   = tid >> 2, vseg = tid & 3;   // sV: 64 rows x 4 segs
    const _Float16* ksrc = kT + ((size_t)b * 2048 + kj) * 64 + kseg * 8;
    const _Float16* vsrc = vR + ((size_t)b * 64 + vc) * 2048 + vseg * 8;

    // A fragments of q^T (row l fixed per lane, 64 channels in 2 K-chunks)
    v16h aq[2];
    {
        const _Float16* qp = qT + ((size_t)b * 2048 + l0 + lrow) * 64;
        const int k0 = half ? 8 : 0;
#pragma unroll
        for (int ch = 0; ch < 2; ++ch)
#pragma unroll
            for (int i = 0; i < 8; ++i) {
                aq[ch][i]     = qp[ch * 32 + k0 + i];
                aq[ch][8 + i] = qp[ch * 32 + 16 + k0 + i];
            }
    }

    float mrow[8], srow[8];
#pragma unroll
    for (int r = 0; r < 8; ++r) { mrow[r] = -__builtin_inff(); srow[r] = 0.f; }
    v8f o[4] = {};

    // prologue stage of chunk 0
    cp16_g2l(ksrc, &sK[0][kj][kseg * 8]);
    cp16_g2l(vsrc, &sV[0][vc][vseg * 8]);

    for (int jc = 0; jc < 64; ++jc) {
        const int buf = jc & 1;
        wait_async();
        __syncthreads();                       // staged chunk jc visible to all waves
        if (jc + 1 < 64) {                     // kick off next chunk's DMA
            cp16_g2l(ksrc + (size_t)(jc + 1) * 32 * 64, &sK[buf ^ 1][kj][kseg * 8]);
            cp16_g2l(vsrc + (jc + 1) * 32,              &sV[buf ^ 1][vc][vseg * 8]);
        }

        // ---- S = q^T k for two 16-wide j tiles (fragments from LDS) ----
        v8f s0 = {}, s1 = {};
        {
            const int cb = half * 8;
            v16h b00, b01, b10, b11;
#pragma unroll
            for (int i = 0; i < 16; ++i) {
                const int ci = cb + (i & 7) + ((i >> 3) << 4);
                b00[i] = sK[buf][lrow][ci];
                b01[i] = sK[buf][lrow][32 + ci];
                b10[i] = sK[buf][16 + lrow][ci];
                b11[i] = sK[buf][16 + lrow][32 + ci];
            }
            s0 = WMMA_F16(aq[0], b00, s0);
            s0 = WMMA_F16(aq[1], b01, s0);
            s1 = WMMA_F16(aq[0], b10, s1);
            s1 = WMMA_F16(aq[1], b11, s1);
        }

        // ---- online softmax (row stats reduced across the 16-lane half) ----
        float p0[8], p1[8], scale[8];
#pragma unroll
        for (int r = 0; r < 8; ++r) {
            float cm = fmaxf(s0[r], s1[r]);
#pragma unroll
            for (int m = 1; m < 16; m <<= 1) cm = fmaxf(cm, __shfl_xor(cm, m, 32));
            const float nm = fmaxf(mrow[r], cm);
            scale[r] = __expf(mrow[r] - nm);
            mrow[r]  = nm;
            p0[r] = __expf(s0[r] - nm);
            p1[r] = __expf(s1[r] - nm);
            float cs = p0[r] + p1[r];
#pragma unroll
            for (int m = 1; m < 16; m <<= 1) cs += __shfl_xor(cs, m, 32);
            srow[r] = srow[r] * scale[r] + cs;
        }
#pragma unroll
        for (int t = 0; t < 4; ++t)
#pragma unroll
            for (int r = 0; r < 8; ++r) o[t][r] *= scale[r];

        // ---- P: D-layout -> LDS -> A-layout (wave-private, s_wait_dscnt) ----
#pragma unroll
        for (int r = 0; r < 8; ++r) {
            sP[w][half * 8 + r][lrow]      = (_Float16)p0[r];
            sP[w][half * 8 + r][16 + lrow] = (_Float16)p1[r];
        }
        asm volatile("s_wait_dscnt 0" ::: "memory");
        v16h ap;
        {
            const int k0 = half ? 8 : 0;
#pragma unroll
            for (int i = 0; i < 8; ++i) {
                ap[i]     = sP[w][lrow][k0 + i];
                ap[8 + i] = sP[w][lrow][16 + k0 + i];
            }
        }

        // ---- O += P @ v^T over four 16-channel tiles (B fragments from LDS) ----
#pragma unroll
        for (int ct = 0; ct < 4; ++ct) {
            v16h bvf;
#pragma unroll
            for (int i = 0; i < 16; ++i)
                bvf[i] = sV[buf][ct * 16 + lrow][half * 8 + (i & 7) + ((i >> 3) << 4)];
            o[ct] = WMMA_F16(ap, bvf, o[ct]);
        }
        __syncthreads();                       // all waves done with buf before restage
    }

    // ---- epilogue: normalize, gamma, store channels 0..63 ----
    float inv[8];
#pragma unroll
    for (int r = 0; r < 8; ++r) inv[r] = 1.0f / srow[r];
#pragma unroll
    for (int ct = 0; ct < 4; ++ct) {
        const int c = ct * 16 + lrow;
#pragma unroll
        for (int r = 0; r < 8; ++r)
            out[((size_t)b * 128 + c) * 2048 + l0 + half * 8 + r] =
                gamma * o[ct][r] * inv[r];
    }
}

// ---------------------------------------------------------------------------
extern "C" void kernel_launch(void* const* d_in, const int* in_sizes, int n_in,
                              void* d_out, int out_size, void* d_ws, size_t ws_size,
                              hipStream_t stream) {
    const float* x  = (const float*)d_in[0];
    const float* Wv = (const float*)d_in[1];
    const float* bv = (const float*)d_in[2];
    const float* Wq = (const float*)d_in[3];
    const float* bq = (const float*)d_in[4];
    const float* Wk = (const float*)d_in[5];
    const float* bk = (const float*)d_in[6];
    const float* g  = (const float*)d_in[7];
    float* out = (float*)d_out;

    const size_t n = (size_t)8 * 2048 * 64;     // elems per q/k/v plane (f16)
    _Float16* qT = (_Float16*)d_ws;             // [B][L][64]
    _Float16* kT = qT + n;                      // [B][L][64]
    _Float16* vR = kT + n;                      // [B][64][L]

    vqk_kernel<<<128, 256, 0, stream>>>(x, Wv, bv, Wq, bq, Wk, bk, qT, kT, vR, out);
    attn_kernel<<<128, 256, 0, stream>>>(qT, kT, vR, g, out);
}
